// DDGCLTgat_38439957299980
// MI455X (gfx1250) — matrix operand order repaired and checked by reading, not compiled, not run on hardware
//
#include <hip/hip_runtime.h>
#include <hip/hip_bf16.h>
#include <limits.h>

typedef __attribute__((ext_vector_type(16))) _Float16 v16h;
typedef __attribute__((ext_vector_type(8)))  _Float16 v8h;
typedef __attribute__((ext_vector_type(8)))  float    v8f;

#define NODE_DIM 128
#define HC       128
#define QKVS     512
#define IN_EDGE  64
#define OUT_DIM  32
#define HEADS    4

// ---------- small utility kernels ----------

__global__ void k_copy_f32(const float* __restrict__ src, float* __restrict__ dst, int n) {
  int i = blockIdx.x * blockDim.x + threadIdx.x;
  if (i < n) dst[i] = src[i];
}

__global__ void k_fill_f32(float* __restrict__ p, float v, long n) {
  long i = (long)blockIdx.x * blockDim.x + threadIdx.x;
  if (i < n) p[i] = v;
}

__global__ void k_fill_i32(int* __restrict__ p, int v, long n) {
  long i = (long)blockIdx.x * blockDim.x + threadIdx.x;
  if (i < n) p[i] = v;
}

// gather x0 = node_emb[n_id] (fp32 -> f16)
__global__ void k_gather_x(const int* __restrict__ n_id, const float* __restrict__ emb,
                           _Float16* __restrict__ x, int N) {
  long i = (long)blockIdx.x * blockDim.x + threadIdx.x;
  if (i >= (long)N * NODE_DIM) return;
  int node = (int)(i / NODE_DIM), c = (int)(i % NODE_DIM);
  x[i] = (_Float16)emb[(long)n_id[node] * NODE_DIM + c];
}

// edge_attr = concat(edge_feats, cos(t*tw+tb)) -> f16 [E,64]
__global__ void k_time_edge(const float* __restrict__ times, const float* __restrict__ feats,
                            const float* __restrict__ tw, const float* __restrict__ tb,
                            _Float16* __restrict__ ea, int E) {
  long i = (long)blockIdx.x * blockDim.x + threadIdx.x;
  if (i >= (long)E * IN_EDGE) return;
  int e = (int)(i / IN_EDGE), c = (int)(i % IN_EDGE);
  float v;
  if (c < 32) v = feats[(long)e * 32 + c];
  else        v = __cosf(times[e] * tw[c - 32] + tb[c - 32]);
  ea[i] = (_Float16)v;
}

// ---------- B-matrix fragment packing ----------
// Pack fp32 W[K, Ncol] into fragment-major f16: [colTile][kSlab][lane][16 halves]
// so each lane loads its whole WMMA B fragment as one contiguous 32B read.
__global__ void k_pack_b(const float* __restrict__ src, _Float16* __restrict__ dst,
                         int K, int Ncol, int tile0) {
  int slabs = K >> 5;
  long total = (long)(Ncol >> 4) * slabs * 32;
  long i = (long)blockIdx.x * blockDim.x + threadIdx.x;
  if (i >= total) return;
  int lane = (int)(i & 31);
  int s    = (int)((i >> 5) % slabs);
  int t    = (int)((i >> 5) / slabs);
  int n = lane & 15, sel = lane >> 4;
  _Float16* d = dst + (((long)(tile0 + t) * slabs + s) * 32 + lane) * 16;
#pragma unroll
  for (int j = 0; j < 8; ++j) {
    int k = 32 * s + 8 * sel + 2 * j + (j >= 4 ? 8 : 0);
    d[2 * j]     = (_Float16)src[(long)k * Ncol + 16 * t + n];
    d[2 * j + 1] = (_Float16)src[(long)(k + 1) * Ncol + 16 * t + n];
  }
}

// ---------- WMMA GEMM: C[M,NOUT] = A[M,K](f16) * Bpacked + bias ----------
// ISA 7.12.2 A layout: lane<16 holds K in {0..7, 16..23}, lane>=16 {8..15, 24..31}.

__device__ inline v16h load_a_frag(const _Float16* __restrict__ A, int lda,
                                   int row0, int k0, int lane) {
  int m = lane & 15, sel = lane >> 4;
  const _Float16* p = A + (long)(row0 + m) * lda + k0 + sel * 8;
  v16h a;
#pragma unroll
  for (int j = 0; j < 8; ++j)          // p[0..7] and p[16..23]: two b128 loads
    a[j] = p[j];
#pragma unroll
  for (int j = 0; j < 8; ++j)
    a[8 + j] = p[16 + j];
  return a;
}

template <bool OUT_HALF, int K, int NOUT>
__global__ __launch_bounds__(128)
void k_gemm16(const _Float16* __restrict__ A, const _Float16* __restrict__ Bp,
              const float* __restrict__ bias, void* __restrict__ Out, int M) {
  constexpr int SLABS = K / 32;
  int wave = threadIdx.x >> 5, lane = threadIdx.x & 31;
  int row0 = (blockIdx.x * 4 + wave) * 16;
  if (row0 >= M) return;
  int colTile = blockIdx.y;
  int n = lane & 15, sel = lane >> 4;

  float bv = bias ? bias[colTile * 16 + n] : 0.f;
  v8f c;
#pragma unroll
  for (int r = 0; r < 8; ++r) c[r] = bv;

  const v16h* bfr = (const v16h*)Bp + (long)colTile * SLABS * 32 + lane;
#pragma unroll
  for (int s = 0; s < SLABS; ++s) {
    v16h a = load_a_frag(A, K, row0, 32 * s, lane);
    v16h b = bfr[s * 32];              // one contiguous 32B fragment load
    c = __builtin_amdgcn_wmma_f32_16x16x32_f16(false, a, false, b, (short)0, c, false, false);
  }

  // C/D layout: VGPR r -> row = row0 + r + 8*sel, col = colTile*16 + (lane&15)
  long base = (long)(row0 + 8 * sel) * NOUT + colTile * 16 + n;
  if (row0 + 16 <= M) {                // full tile: straight-line stores
    if (OUT_HALF) {
      _Float16* o = (_Float16*)Out + base;
#pragma unroll
      for (int r = 0; r < 8; ++r) o[r * NOUT] = (_Float16)c[r];
    } else {
      float* o = (float*)Out + base;
#pragma unroll
      for (int r = 0; r < 8; ++r) o[r * NOUT] = c[r];
    }
  } else {                             // ragged tail (unused when 16 | M)
#pragma unroll
    for (int r = 0; r < 8; ++r) {
      int row = row0 + r + 8 * sel;
      if (row < M) {
        if (OUT_HALF) ((_Float16*)Out)[base + (long)r * NOUT] = (_Float16)c[r];
        else          ((float*)Out)[base + (long)r * NOUT]    = c[r];
      }
    }
  }
}

// ---------- attention (segment softmax via ordered-int atomic max) ----------

__device__ inline int f_ord(float f) { int i = __float_as_int(f); return i >= 0 ? i : (i ^ 0x7fffffff); }
__device__ inline float f_dec(int i) { return __int_as_float(i >= 0 ? i : (i ^ 0x7fffffff)); }

// pass A: alpha[e,h] = <q[dst,h], k[src,h]+e[e,h]> / sqrt(32); m[dst,h] = max
__global__ void k_attn_scores(const int* __restrict__ ei, const _Float16* __restrict__ qk,
                              const _Float16* __restrict__ eh, float* __restrict__ alpha,
                              int* __restrict__ mbuf, int E) {
  long t = (long)blockIdx.x * blockDim.x + threadIdx.x;
  if (t >= (long)E * HEADS) return;
  int e = (int)(t >> 2), h = (int)(t & 3);
  int s = ei[e], d = ei[(long)E + e];
  const v8h* q8 = (const v8h*)(qk + (long)d * QKVS + h * OUT_DIM);
  const v8h* k8 = (const v8h*)(qk + (long)s * QKVS + HC + h * OUT_DIM);
  const v8h* e8 = (const v8h*)(eh + (long)e * HC + h * OUT_DIM);
  float acc = 0.f;
#pragma unroll
  for (int j = 0; j < 4; ++j) {
    v8h qv = q8[j], kv = k8[j], ev = e8[j];
#pragma unroll
    for (int c = 0; c < 8; ++c)
      acc += (float)qv[c] * ((float)kv[c] + (float)ev[c]);
  }
  float al = acc * 0.17677669529663687f;  // 1/sqrt(32)
  alpha[(long)e * HEADS + h] = al;
  atomicMax(&mbuf[(long)d * HEADS + h], f_ord(al));
}

// pass B: a = exp(alpha-m); s[dst,h] += a; out_acc[dst,h,:] += a*(v[src]+e)
__global__ void k_attn_accum(const int* __restrict__ ei, const _Float16* __restrict__ qk,
                             const _Float16* __restrict__ eh, const float* __restrict__ alpha,
                             const int* __restrict__ mbuf, float* __restrict__ sbuf,
                             float* __restrict__ oacc, int E) {
  long t = (long)blockIdx.x * blockDim.x + threadIdx.x;
  if (t >= (long)E * HEADS) return;
  int e = (int)(t >> 2), h = (int)(t & 3);
  int s = ei[e], d = ei[(long)E + e];
  float mm = f_dec(mbuf[(long)d * HEADS + h]);
  float a = __expf(alpha[(long)e * HEADS + h] - mm);
  atomicAdd(&sbuf[(long)d * HEADS + h], a);
  const v8h* v8 = (const v8h*)(qk + (long)s * QKVS + 2 * HC + h * OUT_DIM);
  const v8h* e8 = (const v8h*)(eh + (long)e * HC + h * OUT_DIM);
  float* op = oacc + (long)d * HC + h * OUT_DIM;
#pragma unroll
  for (int j = 0; j < 4; ++j) {
    v8h vv = v8[j], ev = e8[j];
#pragma unroll
    for (int c = 0; c < 8; ++c)
      atomicAdd(&op[8 * j + c], a * ((float)vv[c] + (float)ev[c]));
  }
}

// combine: x_out = oacc / (s + eps) + skip ; optional relu
__global__ void k_combine(const float* __restrict__ oacc, const float* __restrict__ sbuf,
                          const _Float16* __restrict__ qk, _Float16* __restrict__ xout,
                          int N, int relu) {
  long i = (long)blockIdx.x * blockDim.x + threadIdx.x;
  if (i >= (long)N * HC) return;
  int nno = (int)(i / HC), hc = (int)(i % HC);
  int h = hc >> 5;
  float v = oacc[i] / (sbuf[(long)nno * HEADS + h] + 1e-16f)
          + (float)qk[(long)nno * QKVS + 3 * HC + hc];              // skip block
  if (relu) v = v > 0.f ? v : 0.f;
  xout[i] = (_Float16)v;
}

// ---------- host orchestration ----------

static inline unsigned bl(long n, int b) { return (unsigned)((n + b - 1) / b); }

extern "C" void kernel_launch(void* const* d_in, const int* in_sizes, int n_in,
                              void* d_out, int out_size, void* d_ws, size_t ws_size,
                              hipStream_t stream) {
  const int*   n_id   = (const int*)d_in[0];
  const int*   ei     = (const int*)d_in[1];
  const float* etimes = (const float*)d_in[2];
  const float* efeats = (const float*)d_in[3];
  const float* emb    = (const float*)d_in[5];
  const float* tw     = (const float*)d_in[6];
  const float* tb     = (const float*)d_in[7];
  const float* W1[9]  = { (const float*)d_in[8],  (const float*)d_in[9],  (const float*)d_in[10],
                          (const float*)d_in[11], (const float*)d_in[12], (const float*)d_in[13],
                          (const float*)d_in[14], (const float*)d_in[15], (const float*)d_in[16] };
  const float* W2[9]  = { (const float*)d_in[17], (const float*)d_in[18], (const float*)d_in[19],
                          (const float*)d_in[20], (const float*)d_in[21], (const float*)d_in[22],
                          (const float*)d_in[23], (const float*)d_in[24], (const float*)d_in[25] };
  const float* pW = (const float*)d_in[26];
  const float* pb = (const float*)d_in[27];

  const int N = in_sizes[0];
  const int E = in_sizes[2];
  const int batch = out_size / OUT_DIM;

  // carve workspace
  char* ws = (char*)d_ws; size_t off = 0;
  auto carve = [&](size_t bytes) -> void* {
    void* p = ws + off; off += (bytes + 255) & ~(size_t)255; return p;
  };
  _Float16* xA   = (_Float16*)carve((size_t)N * NODE_DIM * 2);
  _Float16* xB   = (_Float16*)carve((size_t)N * NODE_DIM * 2);
  _Float16* qk   = (_Float16*)carve((size_t)N * QKVS * 2);
  _Float16* eaH  = (_Float16*)carve((size_t)E * IN_EDGE * 2);
  _Float16* eH   = (_Float16*)carve((size_t)E * HC * 2);
  float*    alpha= (float*)carve((size_t)E * HEADS * 4);
  int*      mbuf = (int*)  carve((size_t)N * HEADS * 4);
  float*    sbuf = (float*)carve((size_t)N * HEADS * 4);
  float*    oacc = (float*)carve((size_t)N * HC * 4);
  _Float16* Wcat = (_Float16*)carve((size_t)NODE_DIM * QKVS * 2);  // packed
  float*    bcat = (float*)carve(QKVS * 4);
  _Float16* eWp  = (_Float16*)carve((size_t)IN_EDGE * HC * 2);     // packed
  _Float16* pWp  = (_Float16*)carve((size_t)HC * OUT_DIM * 2);     // packed
  (void)ws_size; (void)n_in;

  // stage 0: inputs
  k_gather_x<<<bl((long)N * NODE_DIM, 256), 256, 0, stream>>>(n_id, emb, xA, N);
  k_time_edge<<<bl((long)E * IN_EDGE, 256), 256, 0, stream>>>(etimes, efeats, tw, tb, eaH, E);

  auto conv = [&](const _Float16* xin, _Float16* xout, const float* const* W, int relu) {
    // pack [Wq|Wk|Wv|Ws] into fragment-major f16 (8 col-tiles each, K=128)
    long packthr = (long)(HC / 16) * (NODE_DIM / 32) * 32;
    for (int p = 0; p < 4; ++p) {
      const float* w = (p < 3) ? W[2 * p] : W[7];
      const float* b = (p < 3) ? W[2 * p + 1] : W[8];
      k_pack_b<<<bl(packthr, 256), 256, 0, stream>>>(w, Wcat, NODE_DIM, HC, p * (HC / 16));
      k_copy_f32<<<bl(HC, 256), 256, 0, stream>>>(b, bcat + p * HC, HC);
    }
    long packthrE = (long)(HC / 16) * (IN_EDGE / 32) * 32;
    k_pack_b<<<bl(packthrE, 256), 256, 0, stream>>>(W[6], eWp, IN_EDGE, HC, 0);

    // q|k|v|skip GEMM  (N x 128 x 512)
    dim3 g1(bl(((long)N + 15) / 16, 4), QKVS / 16);
    k_gemm16<true, NODE_DIM, QKVS><<<g1, 128, 0, stream>>>(xin, Wcat, bcat, qk, N);
    // edge GEMM  (E x 64 x 128)
    dim3 g2(bl(((long)E + 15) / 16, 4), HC / 16);
    k_gemm16<true, IN_EDGE, HC><<<g2, 128, 0, stream>>>(eaH, eWp, nullptr, eH, E);

    // attention
    k_fill_i32<<<bl((long)N * HEADS, 256), 256, 0, stream>>>(mbuf, INT_MIN, (long)N * HEADS);
    k_fill_f32<<<bl((long)N * HEADS, 256), 256, 0, stream>>>(sbuf, 0.f, (long)N * HEADS);
    k_fill_f32<<<bl((long)N * HC, 256), 256, 0, stream>>>(oacc, 0.f, (long)N * HC);
    k_attn_scores<<<bl((long)E * HEADS, 256), 256, 0, stream>>>(ei, qk, eH, alpha, mbuf, E);
    k_attn_accum<<<bl((long)E * HEADS, 256), 256, 0, stream>>>(ei, qk, eH, alpha, mbuf, sbuf, oacc, E);
    k_combine<<<bl((long)N * HC, 256), 256, 0, stream>>>(oacc, sbuf, qk, xout, N, relu);
  };

  conv(xA, xB, W1, 1);
  conv(xB, xA, W2, 0);

  // projector: out[batch,32] = x2[:batch] @ pW + pb  (f32 out)
  long packthrP = (long)(OUT_DIM / 16) * (HC / 32) * 32;
  k_pack_b<<<bl(packthrP, 256), 256, 0, stream>>>(pW, pWp, HC, OUT_DIM, 0);
  dim3 g3(bl(((long)batch + 15) / 16, 4), OUT_DIM / 16);
  k_gemm16<false, HC, OUT_DIM><<<g3, 128, 0, stream>>>(xA, pWp, pb, d_out, batch);
}
